// CHGAT_34634616274997
// MI455X (gfx1250) — matrix operand order repaired and checked by reading, not compiled, not run on hardware
//
#include <hip/hip_runtime.h>
#include <hip/hip_bf16.h>

// ---------------------------------------------------------------------------
// CHGAT forward for MI455X (gfx1250, wave32).
// All dense GEMMs go through v_wmma_f32_16x16x32_f16 (f16 in, f32 accum).
// Weights are pre-packed into WMMA B-fragment layout so each fragment load is
// one aligned 32B vector load (2x global_load_b128) instead of 16x u16 loads.
// GAT edge-softmax uses ordered-uint atomicMax + f32 atomicAdd scatter.
// ---------------------------------------------------------------------------

typedef __attribute__((ext_vector_type(16))) _Float16 v16h;
typedef __attribute__((ext_vector_type(8)))  _Float16 v8h;
typedef __attribute__((ext_vector_type(8)))  float    v8f;
typedef __attribute__((ext_vector_type(4)))  float    v4f;

#define DEVFN __device__ __forceinline__

constexpr int N_   = 65536;
constexpr int B_   = 128;
constexpr int E_   = 65536;
constexpr int IN_  = 512;
constexpr int HID_ = 64;
constexpr int OUT_ = 64;
constexpr int H_   = 4;
constexpr int NC_  = 4;
constexpr int ETOT = 2 * E_ + N_;      // TD + BU(flipped) + self loops = 196608

// ---------------------------- WMMA helpers ---------------------------------

DEVFN v8f wmma_f16(v16h a, v16h b, v8f c) {
  // 8 args: (neg_a, A, neg_b, B, c_mod, C, reuse_a, reuse_b)
  return __builtin_amdgcn_wmma_f32_16x16x32_f16(false, a, false, b, (short)0, c,
                                                false, false);
}

// A fragment, 16x32 f16 tile of row-major [rows, lda] matrix.
// ISA layout: lane m=l&15 is row; ko=(l<16?0:8); halves j: K = ko+(j&7)+(j>=8?16:0)
// => two contiguous runs of 8 halves -> 16-byte vector loads.
DEVFN v16h load_a_f16(const _Float16* __restrict__ A, int lda, int row0, int k0,
                      int lane) {
  int m  = lane & 15;
  int ko = (lane >> 4) << 3;
  const _Float16* p = A + (size_t)(row0 + m) * lda + k0 + ko;
  v8h lo = *(const v8h*)(p);
  v8h hi = *(const v8h*)(p + 16);
  v16h a;
#pragma unroll
  for (int j = 0; j < 8; ++j) { a[j] = lo[j]; a[j + 8] = hi[j]; }
  return a;
}

// Same fragment but sourced from f32 memory with convert (for raw input x).
DEVFN v16h load_a_f32(const float* __restrict__ A, int lda, int row0, int k0,
                      int lane) {
  int m  = lane & 15;
  int ko = (lane >> 4) << 3;
  const float* p = A + (size_t)(row0 + m) * lda + k0 + ko;
  v4f p0 = *(const v4f*)(p);
  v4f p1 = *(const v4f*)(p + 4);
  v4f p2 = *(const v4f*)(p + 16);
  v4f p3 = *(const v4f*)(p + 20);
  v16h a;
#pragma unroll
  for (int j = 0; j < 4; ++j) {
    a[j]      = (_Float16)p0[j];
    a[j + 4]  = (_Float16)p1[j];
    a[j + 8]  = (_Float16)p2[j];
    a[j + 12] = (_Float16)p3[j];
  }
  return a;
}

// Packed-B fragment: lane's 16 halves stored contiguously (32B, aligned).
// Packed index: frag = nt * (K/32) + kt; element = (frag*32 + lane)*16 + j.
DEVFN v16h load_b_packed(const _Float16* __restrict__ P, int numKt, int kt,
                         int nt, int lane) {
  return *((const v16h*)P + ((size_t)(nt * numKt + kt) * 32 + lane));
}

// Ordered-uint encoding so unsigned atomicMax == float max (handles negatives).
DEVFN unsigned ordEnc(float f) {
  unsigned u = __float_as_uint(f);
  return (u & 0x80000000u) ? ~u : (u | 0x80000000u);
}
DEVFN float ordDec(unsigned u) {
  unsigned b = (u & 0x80000000u) ? (u & 0x7fffffffu) : ~u;
  return __uint_as_float(b);
}

DEVFN float sigmoidf_(float z) { return 1.f / (1.f + __expf(-z)); }

// merged edge list: [0,E): (s,d); [E,2E): flipped; [2E,2E+N): self loops
DEVFN void edge_sd(const int* __restrict__ ei, int e, int& s, int& d) {
  if (e < E_)            { s = ei[e];           d = ei[E_ + e]; }
  else if (e < 2 * E_)   { int j = e - E_; s = ei[E_ + j]; d = ei[j]; }
  else                   { int v = e - 2 * E_; s = v; d = v; }
}

// ------------------------------ utility kernels -----------------------------

__global__ void k_fill_f32(float* __restrict__ p, float v, int n) {
  int i = blockIdx.x * blockDim.x + threadIdx.x;
  if (i < n) p[i] = v;
}
__global__ void k_fill_u32(unsigned* __restrict__ p, unsigned v, int n) {
  int i = blockIdx.x * blockDim.x + threadIdx.x;
  if (i < n) p[i] = v;
}

// Pack f32 row-major weight [K, Ncols] into f16 WMMA B-fragment layout.
// One thread per (fragment, lane); strided gather once, contiguous forever.
__global__ void k_packB(const float* __restrict__ W, _Float16* __restrict__ P,
                        int K, int Ncols) {
  int idx = blockIdx.x * blockDim.x + threadIdx.x;
  int numKt = K >> 5, numNt = Ncols >> 4;
  if (idx >= numKt * numNt * 32) return;
  int lane = idx & 31;
  int frag = idx >> 5;                    // nt * numKt + kt
  int kt = frag % numKt, nt = frag / numKt;
  int n  = nt * 16 + (lane & 15);
  int kb = kt * 32 + ((lane >> 4) << 3);  // lanes 0-15: K 0..7,16..23, lanes 16-31: K 8..15,24..31
  _Float16* o = P + ((size_t)frag * 32 + lane) * 16;
#pragma unroll
  for (int j = 0; j < 16; ++j) o[j] = (_Float16)W[(size_t)(kb + (j & 7) + ((j >> 3) << 4)) * Ncols + n];
}

// hc1 = root @ W_claim1, hc2 = root @ W_claim2  (B=128 rows only)
__global__ void k_root(const float* __restrict__ root,
                       const float* __restrict__ Wc1, const float* __restrict__ Wc2,
                       float* __restrict__ hc1, float* __restrict__ hc2) {
  int b = blockIdx.x, j = threadIdx.x;
  float a1 = 0.f, a2 = 0.f;
  for (int k = 0; k < IN_; ++k) {
    float r = root[(size_t)b * IN_ + k];
    a1 += r * Wc1[(size_t)k * HID_ + j];
    a2 += r * Wc2[(size_t)k * HID_ + j];
  }
  hc1[b * HID_ + j] = a1;
  hc2[b * HID_ + j] = a2;
}

// -------- Stage 1: h_x = x @ W_post1 (WMMA), fused sigmoid-gate epilogue ----
// h1[n, 0:64]  = g*h_x + (1-g)*h_c,  h1[n, 64:128] = h_x   (stored f16)
__global__ __launch_bounds__(256) void k_gate1(const float* __restrict__ x,
                                               const _Float16* __restrict__ Wp,
                                               const float* __restrict__ hc1,
                                               const int* __restrict__ batch,
                                               _Float16* __restrict__ h1h) {
  int lane = threadIdx.x & 31;
  int row0 = (blockIdx.x * 8 + (threadIdx.x >> 5)) * 16;
  const int numKt = IN_ >> 5;             // 16
  v8f acc[4];
#pragma unroll
  for (int t = 0; t < 4; ++t) acc[t] = (v8f)0.f;
  for (int kt = 0; kt < numKt; ++kt) {
    v16h a = load_a_f32(x, IN_, row0, kt << 5, lane);
#pragma unroll
    for (int t = 0; t < 4; ++t)
      acc[t] = wmma_f16(a, load_b_packed(Wp, numKt, kt, t, lane), acc[t]);
  }
  int n = lane & 15, mo = (lane >> 4) * 8;
#pragma unroll
  for (int t = 0; t < 4; ++t) {
#pragma unroll
    for (int r = 0; r < 8; ++r) {
      int row = row0 + mo + r;
      int col = t * 16 + n;
      float hx = acc[t][r];
      float hc = hc1[batch[row] * HID_ + col];
      float g  = sigmoidf_(hx + hc);
      h1h[(size_t)row * 128 + col]      = (_Float16)(g * hx + (1.f - g) * hc);
      h1h[(size_t)row * 128 + 64 + col] = (_Float16)hx;
    }
  }
}

// ---------------- generic f16 WMMA GEMM: wave = 16 rows x 64 cols -----------
// C[z] = A[z] @ packedB, row-major f32 out. grid.x: 128 rows/block,
// grid.y: col-tiles of 64, grid.z: batch (per-head) with element strides.
__global__ __launch_bounds__(256) void k_gemm16(const _Float16* __restrict__ A,
                                                int lda, long aB,
                                                const _Float16* __restrict__ Pb,
                                                float* __restrict__ C,
                                                int ldc, long cB, int K) {
  int lane = threadIdx.x & 31;
  int row0 = (blockIdx.x * 8 + (threadIdx.x >> 5)) * 16;
  int nt0  = blockIdx.y * 4;
  const int numKt = K >> 5;
  const _Float16* Ap = A + (size_t)blockIdx.z * aB;
  float*          Cp = C + (size_t)blockIdx.z * cB;
  v8f a0 = (v8f)0.f, a1 = (v8f)0.f, a2 = (v8f)0.f, a3 = (v8f)0.f;
  for (int kt = 0; kt < numKt; ++kt) {
    v16h a = load_a_f16(Ap, lda, row0, kt << 5, lane);
    a0 = wmma_f16(a, load_b_packed(Pb, numKt, kt, nt0 + 0, lane), a0);
    a1 = wmma_f16(a, load_b_packed(Pb, numKt, kt, nt0 + 1, lane), a1);
    a2 = wmma_f16(a, load_b_packed(Pb, numKt, kt, nt0 + 2, lane), a2);
    a3 = wmma_f16(a, load_b_packed(Pb, numKt, kt, nt0 + 3, lane), a3);
  }
  int n = lane & 15, mo = (lane >> 4) * 8;
  int n0 = nt0 * 16;
#pragma unroll
  for (int r = 0; r < 8; ++r) {
    size_t base = (size_t)(row0 + mo + r) * ldc + n0 + n;
    Cp[base +  0] = a0[r];
    Cp[base + 16] = a1[r];
    Cp[base + 32] = a2[r];
    Cp[base + 48] = a3[r];
  }
}

// ------------------------- GAT attention (scatter) --------------------------

__global__ void k_att_scores(const float* __restrict__ hf,
                             const float* __restrict__ aws,
                             const float* __restrict__ awd,
                             float* __restrict__ asrc, float* __restrict__ adst) {
  int idx = blockIdx.x * blockDim.x + threadIdx.x;
  if (idx >= N_ * H_) return;
  int n = idx >> 2, h = idx & 3;
  const v4f* row = (const v4f*)(hf + (size_t)n * 256 + h * 64);
  const v4f* ws  = (const v4f*)(aws + h * 64);
  const v4f* wd  = (const v4f*)(awd + h * 64);
  float s = 0.f, d = 0.f;
#pragma unroll
  for (int c = 0; c < 16; ++c) {
    v4f v = row[c], a = ws[c], b = wd[c];
#pragma unroll
    for (int j = 0; j < 4; ++j) { s += v[j] * a[j]; d += v[j] * b[j]; }
  }
  asrc[idx] = s;
  adst[idx] = d;
}

__global__ void k_edge_lm(const int* __restrict__ ei, const float* __restrict__ asrc,
                          const float* __restrict__ adst, float* __restrict__ ebuf,
                          unsigned* __restrict__ mord) {
  int idx = blockIdx.x * blockDim.x + threadIdx.x;
  if (idx >= ETOT * H_) return;
  int e = idx >> 2, h = idx & 3;
  int s, d; edge_sd(ei, e, s, d);
  float v = asrc[s * 4 + h] + adst[d * 4 + h];
  v = v > 0.f ? v : 0.2f * v;                       // LeakyReLU(0.2)
  ebuf[idx] = v;
  atomicMax(&mord[d * 4 + h], ordEnc(v));
}

__global__ void k_edge_ed(const int* __restrict__ ei, float* __restrict__ ebuf,
                          const unsigned* __restrict__ mord, float* __restrict__ den) {
  int idx = blockIdx.x * blockDim.x + threadIdx.x;
  if (idx >= ETOT * H_) return;
  int e = idx >> 2, h = idx & 3;
  int s, d; edge_sd(ei, e, s, d);
  float m  = ordDec(mord[d * 4 + h]);
  float ex = __expf(ebuf[idx] - m);
  ebuf[idx] = ex;
  atomicAdd(&den[d * 4 + h], ex);
}

__global__ __launch_bounds__(256) void k_edge_agg(const int* __restrict__ ei,
                                                  const float* __restrict__ ebuf,
                                                  const float* __restrict__ den,
                                                  const float* __restrict__ hf,
                                                  float* __restrict__ agg) {
  int e = blockIdx.x, t = threadIdx.x, h = t >> 6;
  int s, d; edge_sd(ei, e, s, d);
  float alpha = ebuf[e * 4 + h] / den[d * 4 + h];
  atomicAdd(&agg[(size_t)d * 256 + t], alpha * hf[(size_t)s * 256 + t]);
}

// ----------------------------- epilogue kernels -----------------------------

__global__ void k_gat1_post(const float* __restrict__ agg, const float* __restrict__ b1,
                            _Float16* __restrict__ xrh) {
  int idx = blockIdx.x * blockDim.x + threadIdx.x;
  if (idx >= N_ * 256) return;
  float v = agg[idx] + b1[idx & 255];
  xrh[idx] = (_Float16)fmaxf(v, 0.f);
}

__global__ void k_x3(const float* __restrict__ hs, const float* __restrict__ hc2,
                     const int* __restrict__ batch, _Float16* __restrict__ x3h) {
  int idx = blockIdx.x * blockDim.x + threadIdx.x;
  if (idx >= N_ * 256) return;
  int n = idx >> 8, j = idx & 255, h = j >> 6, c = j & 63;
  float hv = hs[idx];
  float hc = hc2[batch[n] * HID_ + c];
  float g  = sigmoidf_(hv + hc);
  x3h[(size_t)n * 512 + h * 128 + c]      = (_Float16)(g * hv + (1.f - g) * hc);
  x3h[(size_t)n * 512 + h * 128 + 64 + c] = (_Float16)hv;
}

__global__ void k_gat2_post(const float* __restrict__ agg, const float* __restrict__ b2,
                            float* __restrict__ xr2) {
  int idx = blockIdx.x * blockDim.x + threadIdx.x;
  if (idx >= N_ * 64) return;
  int n = idx >> 6, c = idx & 63;
  const float* a = agg + (size_t)n * 256;
  float v = 0.25f * (a[c] + a[64 + c] + a[128 + c] + a[192 + c]) + b2[c];
  xr2[idx] = fmaxf(v, 0.f);
}

__global__ void k_hjoint(const float* __restrict__ xr2, const int* __restrict__ batch,
                         const int* __restrict__ ptr, _Float16* __restrict__ hjh) {
  int idx = blockIdx.x * blockDim.x + threadIdx.x;
  if (idx >= N_ * 64) return;
  int n = idx >> 6, c = idx & 63;
  int rnode  = ptr[batch[n]];
  float hc3  = xr2[(size_t)rnode * 64 + c];
  float v    = xr2[idx];
  _Float16* o = hjh + (size_t)n * 256;
  o[c]        = (_Float16)hc3;
  o[64 + c]   = (_Float16)v;
  o[128 + c]  = (_Float16)(v * hc3);
  o[192 + c]  = (_Float16)fabsf(hc3 - v);
}

// one wave per node: tanh -> row softmax(64) -> scatter into per-graph sums
__global__ __launch_bounds__(256) void k_beta(const float* __restrict__ tfc,
                                              const float* __restrict__ bfc,
                                              const float* __restrict__ xr2,
                                              const int* __restrict__ batch,
                                              float* __restrict__ shat,
                                              float* __restrict__ sacc) {
  int lane = threadIdx.x & 31;
  int node = blockIdx.x * 8 + (threadIdx.x >> 5);
  float v0 = tanhf(tfc[(size_t)node * 64 + lane]      + bfc[lane]);
  float v1 = tanhf(tfc[(size_t)node * 64 + 32 + lane] + bfc[32 + lane]);
  float mx = fmaxf(v0, v1);
#pragma unroll
  for (int o = 16; o; o >>= 1) mx = fmaxf(mx, __shfl_xor(mx, o, 32));
  float e0 = __expf(v0 - mx), e1 = __expf(v1 - mx);
  float sm = e0 + e1;
#pragma unroll
  for (int o = 16; o; o >>= 1) sm += __shfl_xor(sm, o, 32);
  int b = batch[node];
  float x0 = xr2[(size_t)node * 64 + lane];
  float x1 = xr2[(size_t)node * 64 + 32 + lane];
  atomicAdd(&shat[b * 64 + lane],      (e0 / sm) * x0);
  atomicAdd(&shat[b * 64 + 32 + lane], (e1 / sm) * x1);
  atomicAdd(&sacc[b * 64 + lane],      x0);
  atomicAdd(&sacc[b * 64 + 32 + lane], x1);
}

// per-graph classifier + log_softmax; 1 block of 128 threads, thread = graph
__global__ void k_final(const float* __restrict__ shat, const float* __restrict__ sacc,
                        const float* __restrict__ Wclf, const float* __restrict__ bclf,
                        float* __restrict__ out) {
  int b = threadIdx.x;
  const float inv = 1.f / 512.f;          // exactly G nodes per graph
  float l[NC_];
#pragma unroll
  for (int j = 0; j < NC_; ++j) {
    float acc = bclf[j];
    for (int c = 0; c < 64; ++c) {
      acc += (shat[b * 64 + c] * inv) * Wclf[c * NC_ + j];
      acc += (sacc[b * 64 + c] * inv) * Wclf[(64 + c) * NC_ + j];
    }
    l[j] = acc;
  }
  float m = fmaxf(fmaxf(l[0], l[1]), fmaxf(l[2], l[3]));
  float s = 0.f;
#pragma unroll
  for (int j = 0; j < NC_; ++j) s += __expf(l[j] - m);
  float lse = logf(s);
#pragma unroll
  for (int j = 0; j < NC_; ++j) out[b * NC_ + j] = l[j] - m - lse;
}

// ------------------------------- host driver --------------------------------

static inline int cdiv(int a, int b) { return (a + b - 1) / b; }

extern "C" void kernel_launch(void* const* d_in, const int* in_sizes, int n_in,
                              void* d_out, int out_size, void* d_ws, size_t ws_size,
                              hipStream_t stream) {
  const float* x        = (const float*)d_in[0];
  const float* root     = (const float*)d_in[1];
  const int*   ei       = (const int*)d_in[2];
  const int*   batch    = (const int*)d_in[3];
  const int*   ptr      = (const int*)d_in[4];
  const float* W_post1  = (const float*)d_in[5];
  const float* W_claim1 = (const float*)d_in[6];
  const float* W1       = (const float*)d_in[7];
  const float* att_src1 = (const float*)d_in[8];
  const float* att_dst1 = (const float*)d_in[9];
  const float* b1       = (const float*)d_in[10];
  const float* W_post2  = (const float*)d_in[11];
  const float* W_claim2 = (const float*)d_in[12];
  const float* W2       = (const float*)d_in[13];
  const float* att_src2 = (const float*)d_in[14];
  const float* att_dst2 = (const float*)d_in[15];
  const float* b2       = (const float*)d_in[16];
  const float* W_fc     = (const float*)d_in[17];
  const float* b_fc     = (const float*)d_in[18];
  const float* W_clf    = (const float*)d_in[19];
  const float* b_clf    = (const float*)d_in[20];
  float* out = (float*)d_out;

  // ---- carve workspace (256B aligned bump allocator) ----
  char* w = (char*)d_ws;
  auto alloc = [&](size_t bytes) -> char* {
    char* r = w;
    w += (bytes + 255) & ~(size_t)255;
    return r;
  };
  float*     hc1    = (float*)alloc((size_t)B_ * 64 * 4);
  float*     hc2    = (float*)alloc((size_t)B_ * 64 * 4);
  _Float16*  Wp1p   = (_Float16*)alloc((size_t)IN_ * HID_ * 2);   // packed
  _Float16*  W1p    = (_Float16*)alloc((size_t)128 * 256 * 2);    // packed
  _Float16*  Wp2p   = (_Float16*)alloc((size_t)64 * 64 * 2);      // packed
  _Float16*  W2p    = (_Float16*)alloc((size_t)512 * 256 * 2);    // packed
  _Float16*  Wfcp   = (_Float16*)alloc((size_t)256 * 64 * 2);     // packed
  _Float16*  h1h    = (_Float16*)alloc((size_t)N_ * 128 * 2);     // 16 MB
  float*     hfeat  = (float*)alloc((size_t)N_ * 256 * 4);        // 64 MB (GAT1+GAT2)
  float*     asrc   = (float*)alloc((size_t)N_ * 4 * 4);
  float*     adst   = (float*)alloc((size_t)N_ * 4 * 4);
  unsigned*  mord   = (unsigned*)alloc((size_t)N_ * 4 * 4);
  float*     den    = (float*)alloc((size_t)N_ * 4 * 4);
  float*     ebuf   = (float*)alloc((size_t)ETOT * 4 * 4);        // 3 MB
  float*     agg    = (float*)alloc((size_t)N_ * 256 * 4);        // 64 MB (GAT1+GAT2)
  _Float16*  xrh    = (_Float16*)alloc((size_t)N_ * 256 * 2);     // 32 MB
  float*     hs     = (float*)alloc((size_t)N_ * 256 * 4);        // 64 MB (reused)
  _Float16*  x3h    = (_Float16*)alloc((size_t)N_ * 512 * 2);     // 64 MB
  float*     xr2    = (float*)alloc((size_t)N_ * 64 * 4);         // 16 MB
  float*     shat   = (float*)alloc((size_t)B_ * 64 * 4);
  float*     sacc   = (float*)alloc((size_t)B_ * 64 * 4);
  // aliases into freed `hs` region (hs dead after k_x3)
  _Float16*  hjh = (_Float16*)hs;                                 // N*256 f16 = 32 MB
  float*     tfc = (float*)((char*)hs + (size_t)N_ * 256 * 2);    // N*64 f32 = 16 MB

  // ---- pack weights (f32 -> f16 WMMA fragment layout), once per call ----
  k_packB<<<cdiv((IN_ / 32) * (HID_ / 16) * 32, 256), 256, 0, stream>>>(W_post1, Wp1p, IN_, HID_);
  k_packB<<<cdiv((128 / 32) * (256 / 16) * 32, 256), 256, 0, stream>>>(W1, W1p, 128, 256);
  k_packB<<<cdiv((64 / 32) * (64 / 16) * 32, 256), 256, 0, stream>>>(W_post2, Wp2p, 64, 64);
  k_packB<<<cdiv((512 / 32) * (256 / 16) * 32, 256), 256, 0, stream>>>(W2, W2p, 512, 256);
  k_packB<<<cdiv((256 / 32) * (64 / 16) * 32, 256), 256, 0, stream>>>(W_fc, Wfcp, 256, 64);

  // ---- per-graph root projections (128 rows; replaces N-row broadcast GEMM)
  k_root<<<B_, 64, 0, stream>>>(root, W_claim1, W_claim2, hc1, hc2);

  // ---- stage 1: gated fusion (WMMA GEMM + epilogue) -> h1 f16 [N,128] ----
  k_gate1<<<512, 256, 0, stream>>>(x, Wp1p, hc1, batch, h1h);

  // ---- GAT1 linear: hfeat = h1 @ W1, [N,256] ----
  k_gemm16<<<dim3(512, 4, 1), 256, 0, stream>>>(h1h, 128, 0, W1p, hfeat, 256, 0, 128);
  // ---- GAT1 attention ----
  k_att_scores<<<cdiv(N_ * 4, 256), 256, 0, stream>>>(hfeat, att_src1, att_dst1, asrc, adst);
  k_fill_u32<<<cdiv(N_ * 4, 256), 256, 0, stream>>>(mord, 0u, N_ * 4);
  k_fill_f32<<<cdiv(N_ * 4, 256), 256, 0, stream>>>(den, 0.f, N_ * 4);
  k_fill_f32<<<cdiv(N_ * 256, 256), 256, 0, stream>>>(agg, 0.f, N_ * 256);
  k_edge_lm<<<cdiv(ETOT * 4, 256), 256, 0, stream>>>(ei, asrc, adst, ebuf, mord);
  k_edge_ed<<<cdiv(ETOT * 4, 256), 256, 0, stream>>>(ei, ebuf, mord, den);
  k_edge_agg<<<ETOT, 256, 0, stream>>>(ei, ebuf, den, hfeat, agg);
  k_gat1_post<<<cdiv(N_ * 256, 256), 256, 0, stream>>>(agg, b1, xrh);

  // ---- per-head hs = xr[:,h,:] @ W_post2 (batched over heads via grid.z) ----
  k_gemm16<<<dim3(512, 1, 4), 256, 0, stream>>>(xrh, 256, 64, Wp2p, hs, 256, 64, 64);
  // ---- gate 2 -> x3 f16 [N,512] ----
  k_x3<<<cdiv(N_ * 256, 256), 256, 0, stream>>>(hs, hc2, batch, x3h);

  // ---- GAT2 linear: hfeat = x3 @ W2, [N,256] (dominant GEMM, K=512) ----
  k_gemm16<<<dim3(512, 4, 1), 256, 0, stream>>>(x3h, 512, 0, W2p, hfeat, 256, 0, 512);
  // ---- GAT2 attention ----
  k_att_scores<<<cdiv(N_ * 4, 256), 256, 0, stream>>>(hfeat, att_src2, att_dst2, asrc, adst);
  k_fill_u32<<<cdiv(N_ * 4, 256), 256, 0, stream>>>(mord, 0u, N_ * 4);
  k_fill_f32<<<cdiv(N_ * 4, 256), 256, 0, stream>>>(den, 0.f, N_ * 4);
  k_fill_f32<<<cdiv(N_ * 256, 256), 256, 0, stream>>>(agg, 0.f, N_ * 256);
  k_edge_lm<<<cdiv(ETOT * 4, 256), 256, 0, stream>>>(ei, asrc, adst, ebuf, mord);
  k_edge_ed<<<cdiv(ETOT * 4, 256), 256, 0, stream>>>(ei, ebuf, mord, den);
  k_edge_agg<<<ETOT, 256, 0, stream>>>(ei, ebuf, den, hfeat, agg);
  k_gat2_post<<<cdiv(N_ * 64, 256), 256, 0, stream>>>(agg, b2, xr2);

  // ---- joint features + W_fc GEMM + softmax-weighted pooling ----
  k_hjoint<<<cdiv(N_ * 64, 256), 256, 0, stream>>>(xr2, batch, ptr, hjh);
  k_gemm16<<<dim3(512, 1, 1), 256, 0, stream>>>(hjh, 256, 0, Wfcp, tfc, 64, 0, 256);
  k_fill_f32<<<cdiv(B_ * 64, 256), 256, 0, stream>>>(shat, 0.f, B_ * 64);
  k_fill_f32<<<cdiv(B_ * 64, 256), 256, 0, stream>>>(sacc, 0.f, B_ * 64);
  k_beta<<<N_ / 8, 256, 0, stream>>>(tfc, b_fc, xr2, batch, shat, sacc);

  // ---- classifier + log_softmax -> d_out [B,4] ----
  k_final<<<1, B_, 0, stream>>>(shat, sacc, W_clf, b_clf, out);
}
